// Net_b1_23175643529497
// MI455X (gfx1250) — matrix operand order repaired and checked by reading, not compiled, not run on hardware
//
#include <hip/hip_runtime.h>
#include <hip/hip_fp16.h>

// ---------------------------------------------------------------------------
// GeoConv point-cloud net on MI455X (gfx1250, wave32, WMMA + TDM).
// All GEMMs: v_wmma_f32_16x16x32_f16 (f16 in, f32 accumulate).
// A-panels staged into LDS by the Tensor Data Mover (tensor_load_to_lds,
// TENSORcnt), double-buffered against WMMA compute; B-panels staged
// transposed by the waves. Per-edge linears hoisted to per-node GEMMs.
// BN stats via f32 atomics in GEMM epilogue; BN+ReLU(+k-maxpool) fused.
// ---------------------------------------------------------------------------

#define B_  4
#define N_  2048
#define K_  16
#define P_  (B_ * N_)      // 8192 points
#define E_  (P_ * K_)      // 131072 edges
#define EPS_ 1e-5f

typedef unsigned int u32;
typedef __attribute__((ext_vector_type(16))) _Float16 v16h;
typedef __attribute__((ext_vector_type(8)))  float    v8f;
typedef __attribute__((ext_vector_type(4)))  u32      u32x4;
typedef __attribute__((ext_vector_type(8)))  int      i32x8;
typedef __attribute__((ext_vector_type(4)))  int      i32x4;

union Frag16 { v16h v; unsigned u[8]; };

#if defined(__has_builtin)
#  if __has_builtin(__builtin_amdgcn_tensor_load_to_lds) && \
      __has_builtin(__builtin_amdgcn_s_wait_tensorcnt)
#    define USE_TDM 1
#  endif
#endif
#ifndef USE_TDM
#  define USE_TDM 0
#endif

// ---------------------------------------------------------------------------
// Generic WMMA GEMM:  C[M,Nc] (+bias) = A[M,Ka](f16) x W[Ka,Nc](f16)
// block = 128 thr (4 waves), tile 64x64, K step 32, double-buffered LDS.
// Wave 0 drives the TDM for A-panels; all waves stage B transposed.
// M % 64 == 0, Ka % 32 == 0, Nc % 64 == 0 (caller pads).
// ---------------------------------------------------------------------------
__global__ __launch_bounds__(128)
void gemm_wmma_f16(const _Float16* __restrict__ A, int lda,
                   const _Float16* __restrict__ W,   // [Ka, Nc]
                   const float* __restrict__ bias,   // [Nc] or null
                   float* __restrict__ C, int ldc,   // f32 out or null
                   _Float16* __restrict__ Ch, int ldch, // f16 out or null
                   int M, int Ka, int Nc,
                   float* __restrict__ stats,        // [2*Nc] or null
                   int accum)
{
  __shared__ _Float16 As[2][64 * 32];   // [row][k] row-major, TDM-filled
  __shared__ _Float16 Bs[2][64 * 32];   // transposed: [n][k]
  (void)M;
  const int tid  = threadIdx.x;
  const int wv   = tid >> 5;
  const int lane = tid & 31;
  const int bn0  = blockIdx.x * 64;
  const int bm0  = blockIdx.y * 64;

  // ---- TDM: DMA one A panel (64 rows x 16 dwords, row stride lda/2 dwords)
  auto issueA = [&](int k0, int buf) {
#if USE_TDM
    unsigned long long ga =
        (unsigned long long)(const void*)(A + (size_t)bm0 * lda + k0);
    // ISA aperture rule: LDS flat address low 32 bits == LDS byte address.
    u32 lds = (u32)(size_t)(const void*)&As[buf][0];
    u32 rowDw = (u32)(lda >> 1);
    u32x4 g0;
    g0.x = 1u;                                     // count=1, user descriptor
    g0.y = lds;                                    // lds_addr [63:32]
    g0.z = (u32)ga;                                // global_addr [95:64]
    g0.w = ((u32)(ga >> 32) & 0x01FFFFFFu) | (2u << 30); // addr[56:32] | type=2
    i32x8 g1;
    g1[0] = (int)(2u << 16);                       // data_size = 4 bytes
    g1[1] = (int)((rowDw & 0xFFFFu) << 16);        // tensor_dim0 [15:0]
    g1[2] = (int)((rowDw >> 16) & 0xFFFFu);        // tensor_dim0 hi | dim1 lo=0
    g1[3] = (int)(16u | (16u << 16));              // tensor_dim1 hi (=1M) | tile_dim0=16
    g1[4] = (int)64u;                              // tile_dim1=64, tile_dim2=0
    g1[5] = (int)rowDw;                            // tensor_dim0_stride [31:0]
    g1[6] = 0; g1[7] = 0;
    i32x4 gz = {0, 0, 0, 0};
#  if __has_include(<hip/amd_detail/amd_gfx1250_TDM.h>)
    i32x8 gz8 = {0, 0, 0, 0, 0, 0, 0, 0};
    __builtin_amdgcn_tensor_load_to_lds(g0, g1, gz, gz, gz8, 0);
#  else
    __builtin_amdgcn_tensor_load_to_lds(g0, g1, gz, gz, 0);
#  endif
#else
    // manual fallback (wave 0 cooperative copy)
    for (int i = lane; i < 1024; i += 32) {
      int r = i >> 4, d = i & 15;
      ((unsigned*)&As[buf][0])[r * 16 + d] =
          ((const unsigned*)(A + (size_t)(bm0 + r) * lda + k0))[d];
    }
#endif
  };
  auto waitA = [&]() {
#if USE_TDM
    __builtin_amdgcn_s_wait_tensorcnt(0);
#endif
  };
  // ---- stage B tile transposed: Bs[buf][n][k] = W[k0+k][bn0+n]
  auto stageB = [&](int k0, int buf) {
    for (int i = tid; i < 2048; i += 128) {
      int n = i & 63, k = i >> 6;
      Bs[buf][n * 32 + k] = W[(size_t)(k0 + k) * Nc + bn0 + n];
    }
  };

  v8f acc[4] = {};
  const int nk = Ka >> 5;

  // prologue: fill buffer 0 (TDM overlaps with B staging)
  if (wv == 0) issueA(0, 0);
  stageB(0, 0);
  if (wv == 0) waitA();
  __syncthreads();

  for (int kt = 0; kt < nk; ++kt) {
    const int cur = kt & 1, nxt = cur ^ 1;
    // kick next stage while computing current
    if (kt + 1 < nk) {
      if (wv == 0) issueA((kt + 1) << 5, nxt);
      stageB((kt + 1) << 5, nxt);
      if (kt + 2 < nk)
        __builtin_prefetch(W + (size_t)(((kt + 2) << 5) + (tid >> 2)) * Nc + bn0, 0, 3);
    }

    const _Float16* Asc = &As[cur][0];
    const _Float16* Bsc = &Bs[cur][0];

    // A fragment (16x32 f16): lanes 0-15 M=lane K-base 0, lanes 16-31 K-base 8
    Frag16 af;
    {
      int m   = lane & 15;
      int kbu = (lane < 16) ? 0 : 4;            // dword offset for K-base 0 / 8
      const unsigned* Asu = (const unsigned*)(Asc + (wv * 16 + m) * 32);
#pragma unroll
      for (int j = 0; j < 8; ++j) {
        int ku = (j < 4) ? (kbu + j) : (8 + kbu + (j - 4));  // K pairs 0..7 then 16..23
        af.u[j] = Asu[ku];
      }
    }
    // 4 N-subtiles
#pragma unroll
    for (int ns = 0; ns < 4; ++ns) {
      Frag16 bf;
      int nn = ns * 16 + (lane & 15);
      int ko = (lane < 16) ? 0 : 4;             // dword offset for K-base 0 / 8
      const unsigned* Bsu = (const unsigned*)(Bsc + nn * 32);
#pragma unroll
      for (int j = 0; j < 8; ++j) bf.u[j] = Bsu[(j < 4) ? (ko + j) : (8 + ko + (j - 4))];
      acc[ns] = __builtin_amdgcn_wmma_f32_16x16x32_f16(
          false, af.v, false, bf.v, (short)0, acc[ns], false, false);
    }

    if (kt + 1 < nk && wv == 0) waitA();   // next A panel landed
    __syncthreads();                       // B ds-stores + TDM visible to all
  }

  // ---- epilogue: C/D layout lane<16 -> M=r, lane>=16 -> M=8+r
  int mrow = bm0 + wv * 16 + ((lane < 16) ? 0 : 8);
#pragma unroll
  for (int ns = 0; ns < 4; ++ns) {
    int n = bn0 + ns * 16 + (lane & 15);
    float bv = bias ? bias[n] : 0.f;
    float s1 = 0.f, s2 = 0.f;
#pragma unroll
    for (int r = 0; r < 8; ++r) {
      int row = mrow + r;
      float v = acc[ns][r] + bv;
      if (accum) v += C[(size_t)row * ldc + n];
      if (C)  C[(size_t)row * ldc + n] = v;
      if (Ch) Ch[(size_t)row * ldch + n] = (_Float16)v;
      s1 += v; s2 += v * v;
    }
    if (stats) {
      atomicAdd(&stats[n], s1);
      atomicAdd(&stats[Nc + n], s2);
    }
  }
}

// ---------------------------------------------------------------------------
// Brute-force kNN (K=16, incl. self) within each batch; register top-16.
// ---------------------------------------------------------------------------
__global__ __launch_bounds__(256)
void knn_kernel(const float* __restrict__ pos, int* __restrict__ idx,
                float* __restrict__ d2o)
{
  __shared__ float sx[256], sy[256], sz[256];
  int b  = blockIdx.x >> 3;
  int gq = b * N_ + (blockIdx.x & 7) * 256 + threadIdx.x;
  float px = pos[gq * 3], py = pos[gq * 3 + 1], pz = pos[gq * 3 + 2];
  float bd[16]; int bi[16];
#pragma unroll
  for (int s = 0; s < 16; ++s) { bd[s] = 3.0e38f; bi[s] = gq; }

  for (int c0 = 0; c0 < N_; c0 += 256) {
    __syncthreads();
    int ci = b * N_ + c0 + threadIdx.x;
    sx[threadIdx.x] = pos[ci * 3];
    sy[threadIdx.x] = pos[ci * 3 + 1];
    sz[threadIdx.x] = pos[ci * 3 + 2];
    __syncthreads();
    for (int j = 0; j < 256; ++j) {
      float dx = px - sx[j], dy = py - sy[j], dz = pz - sz[j];
      float d2 = dx * dx + dy * dy + dz * dz;
      if (d2 < bd[15]) {
        float dc = d2; int ic = b * N_ + c0 + j;
#pragma unroll
        for (int s = 0; s < 16; ++s) {
          if (dc < bd[s]) {
            float tf = bd[s]; bd[s] = dc; dc = tf;
            int   ti = bi[s]; bi[s] = ic; ic = ti;
          }
        }
      }
    }
  }
#pragma unroll
  for (int s = 0; s < 16; ++s) {
    idx[(size_t)gq * 16 + s] = bi[s];
    d2o[(size_t)gq * 16 + s] = bd[s];
  }
}

// ---------------------------------------------------------------------------
// Small helpers
// ---------------------------------------------------------------------------
__global__ void zero_f32(float* p, int n)
{ int i = blockIdx.x * 256 + threadIdx.x; if (i < n) p[i] = 0.f; }

__global__ void convert_w_f16(const float* __restrict__ w, _Float16* __restrict__ o,
                              int kin, int Nc, long total)
{
  long i = (long)blockIdx.x * 256 + threadIdx.x;
  if (i >= total) return;
  long k = i / (unsigned)Nc; int n = (int)(i - k * Nc);
  o[i] = (k < (long)kin) ? (_Float16)w[k * Nc + n] : (_Float16)0.f;
}

// Wl [6, ci, 64] -> packed [ci, 384] (branch-major columns: j*64+u)
__global__ void pack_wl_f16(const float* __restrict__ wl, _Float16* __restrict__ o, int ci)
{
  long i = (long)blockIdx.x * 256 + threadIdx.x;
  long total = (long)ci * 384;
  if (i >= total) return;
  int c = (int)(i / 384); int j64 = (int)(i - (long)c * 384);
  int j = j64 >> 6, u = j64 & 63;
  o[i] = (_Float16)wl[((size_t)j * ci + c) * 64 + u];
}

// e1 = [pos[s]-pos[t], nrm[s], 0-pad] -> [E,32] f16
__global__ void build_edge1(const float* __restrict__ pos, const float* __restrict__ nrm,
                            const int* __restrict__ idx, _Float16* __restrict__ e1)
{
  int e = blockIdx.x * 256 + threadIdx.x;
  if (e >= E_) return;
  int t = e >> 4, s = idx[e];
  _Float16* o = e1 + (size_t)e * 32;
  o[0] = (_Float16)(pos[s * 3 + 0] - pos[t * 3 + 0]);
  o[1] = (_Float16)(pos[s * 3 + 1] - pos[t * 3 + 1]);
  o[2] = (_Float16)(pos[s * 3 + 2] - pos[t * 3 + 2]);
  o[3] = (_Float16)nrm[s * 3 + 0];
  o[4] = (_Float16)nrm[s * 3 + 1];
  o[5] = (_Float16)nrm[s * 3 + 2];
#pragma unroll
  for (int j = 6; j < 32; ++j) o[j] = (_Float16)0.f;
}

// pn = [pos, nrm, 0-pad] -> [P,32] f16
__global__ void build_pn(const float* __restrict__ pos, const float* __restrict__ nrm,
                         _Float16* __restrict__ o)
{
  int p = blockIdx.x * 256 + threadIdx.x;
  if (p >= P_) return;
  _Float16* t = o + (size_t)p * 32;
  t[0] = (_Float16)pos[p * 3 + 0]; t[1] = (_Float16)pos[p * 3 + 1]; t[2] = (_Float16)pos[p * 3 + 2];
  t[3] = (_Float16)nrm[p * 3 + 0]; t[4] = (_Float16)nrm[p * 3 + 1]; t[5] = (_Float16)nrm[p * 3 + 2];
#pragma unroll
  for (int j = 6; j < 32; ++j) t[j] = (_Float16)0.f;
}

// edge diff: out[e,c] = x[tid,c] - x[sid,c]  (C = 1<<shc)
__global__ void build_diff(const float* __restrict__ x, const int* __restrict__ idx,
                           _Float16* __restrict__ out, int shc)
{
  long i = (long)blockIdx.x * 256 + threadIdx.x;   // over E*C
  int C = 1 << shc;
  long e = i >> shc;
  if (e >= E_) return;
  int c = (int)(i & (C - 1));
  int t = (int)(e >> 4), s = idx[e];
  out[i] = (_Float16)(x[(size_t)t * C + c] - x[(size_t)s * C + c]);
}

// BN scale/shift from sums:  scale = g*rstd,  shift = be - mean*scale
__global__ void bn_finalize(const float* __restrict__ stats, int Nc, float invM,
                            const float* __restrict__ g, const float* __restrict__ be,
                            float* __restrict__ scale, float* __restrict__ shift)
{
  int c = blockIdx.x * 64 + threadIdx.x;
  if (c >= Nc) return;
  float mean = stats[c] * invM;
  float var  = stats[Nc + c] * invM - mean * mean;
  float rs   = rsqrtf(var + EPS_);
  float sc   = g[c] * rs;
  scale[c] = sc;
  shift[c] = be[c] - mean * sc;
}

// elementwise y = relu(scale*x + shift), in place + optional f16 mirror
__global__ __launch_bounds__(256)
void bn_apply_relu(float* __restrict__ x, const float* __restrict__ scale,
                   const float* __restrict__ shift, _Float16* __restrict__ xh,
                   int C, int ldo, int ooff, long total)
{
  long i = (long)blockIdx.x * 256 + threadIdx.x;
  if (i >= total) return;
  long p = i / (unsigned)C;
  int  c = (int)(i - p * C);
  float v = fmaxf(scale[c] * x[i] + shift[c], 0.f);
  x[i] = v;
  if (xh) xh[p * ldo + ooff + c] = (_Float16)v;
}

// PointPlus pool: x[p,c] = max_k relu(scale*h[(p*16+k),c]+shift)
__global__ __launch_bounds__(256)
void pp_pool(const _Float16* __restrict__ h, const float* __restrict__ scale,
             const float* __restrict__ shift, float* __restrict__ xo,
             _Float16* __restrict__ xoh, int shc, int ldo, int ooff)
{
  long i = (long)blockIdx.x * 256 + threadIdx.x;   // over P*C
  int C = 1 << shc;
  long p = i >> shc;
  if (p >= P_) return;
  int c = (int)(i & (C - 1));
  float sc = scale[c], sh = shift[c];
  float m = -3.0e38f;
  const _Float16* hp = h + (((size_t)p << 4) << shc) + c;
#pragma unroll
  for (int k = 0; k < 16; ++k) {
    float v = fmaxf(sc * (float)hp[(size_t)k << shc] + sh, 0.f);
    m = fmaxf(m, v);
  }
  if (xo)  xo[i] = m;
  if (xoh) xoh[p * ldo + ooff + c] = (_Float16)m;
}

// GeoConv edge combine: per target point, distance-weighted sum over k of
// pcos-weighted hi/lo branch projections (gathered from node-level proj).
__global__ __launch_bounds__(64)
void geo_combine(const float* __restrict__ proj,   // [P,384] f32 (bias included)
                 const float* __restrict__ pos, const int* __restrict__ idx,
                 const float* __restrict__ d2, float r2,
                 float* __restrict__ y, float* __restrict__ stats)
{
  __shared__ int   ssid[16];
  __shared__ float spc[16][3];
  __shared__ int   soff[16][3];
  __shared__ float sd[16];
  __shared__ float swn[16];
  int t = blockIdx.x;
  int k = threadIdx.x;
  if (k < 16) {
    int e  = t * 16 + k;
    int s0 = idx[t * 16];
    int s  = (d2[e] <= r2) ? idx[e] : s0;        // radius graph fallback -> nearest
    float dx = pos[s * 3 + 0] - pos[t * 3 + 0];
    float dy = pos[s * 3 + 1] - pos[t * 3 + 1];
    float dz = pos[s * 3 + 2] - pos[t * 3 + 2];
    float dd = dx * dx + dy * dy + dz * dz;
    float dist = sqrtf(fmaxf(dd, 1e-32f));
    float inv = 1.f / dist;
    float cx = dx * inv, cy = dy * inv, cz = dz * inv;
    spc[k][0] = cx * cx; spc[k][1] = cy * cy; spc[k][2] = cz * cz;
    soff[k][0] = 0   + (dx > 0.f ? 64 : 0);
    soff[k][1] = 128 + (dy > 0.f ? 64 : 0);
    soff[k][2] = 256 + (dz > 0.f ? 64 : 0);
    ssid[k] = s;
    sd[k]   = dist;
  }
  __syncthreads();
  if (k < 16) {
    float mx = sd[0];
#pragma unroll
    for (int j = 1; j < 16; ++j) mx = fmaxf(mx, sd[j]);
    float sw = 0.f;
#pragma unroll
    for (int j = 0; j < 16; ++j) { float wj = mx * 1.1f - sd[j]; sw += wj * wj; }
    float w = mx * 1.1f - sd[k];
    swn[k] = (w * w) / sw;
  }
  __syncthreads();
  int c = threadIdx.x;                            // 0..63 channel
  float acc = 0.f;
  for (int kk = 0; kk < 16; ++kk) {
    const float* base = proj + (size_t)ssid[kk] * 384;
    float v = spc[kk][0] * base[soff[kk][0] + c]
            + spc[kk][1] * base[soff[kk][1] + c]
            + spc[kk][2] * base[soff[kk][2] + c];
    acc += swn[kk] * v;
  }
  y[(size_t)t * 64 + c] = acc;
  atomicAdd(&stats[c], acc);
  atomicAdd(&stats[64 + c], acc * acc);
}

// mlpc: global max over N of relu(scale*h+shift), h f16 [B*N,2048]
__global__ __launch_bounds__(256)
void maxpool_bn(const _Float16* __restrict__ h, const float* __restrict__ scale,
                const float* __restrict__ shift, float* __restrict__ xg)
{
  int i = blockIdx.x * 256 + threadIdx.x;         // B*2048
  if (i >= B_ * 2048) return;
  int b = i >> 11, c = i & 2047;
  float sc = scale[c], sh = shift[c];
  const _Float16* p = h + (size_t)b * N_ * 2048 + c;
  float m = -3.0e38f;
  for (int n = 0; n < N_; ++n)
    m = fmaxf(m, fmaxf(sc * (float)p[(size_t)n * 2048] + sh, 0.f));
  xg[i] = m;
}

__global__ void fc1_kernel(const float* __restrict__ xg, const float* __restrict__ W,
                           const float* __restrict__ bias, float* __restrict__ out)
{
  int i = blockIdx.x * 256 + threadIdx.x;         // 4*512
  if (i >= B_ * 512) return;
  int b = i >> 9, o = i & 511;
  const float* xr = xg + (size_t)b * 2048;
  float s = bias[o];
  for (int kk = 0; kk < 2048; ++kk) s += xr[kk] * W[(size_t)kk * 512 + o];
  out[i] = s;
}

__global__ void fc_bn(const float* __restrict__ pre, const float* __restrict__ g,
                      const float* __restrict__ be, float* __restrict__ out)
{
  int o = blockIdx.x * 256 + threadIdx.x;
  if (o >= 512) return;
  float a0 = pre[o], a1 = pre[512 + o], a2 = pre[1024 + o], a3 = pre[1536 + o];
  float m = 0.25f * (a0 + a1 + a2 + a3);
  float v = 0.25f * ((a0 - m) * (a0 - m) + (a1 - m) * (a1 - m)
                   + (a2 - m) * (a2 - m) + (a3 - m) * (a3 - m));
  float sc = g[o] * rsqrtf(v + EPS_);
  float sh = be[o] - m * sc;
  out[o]        = fmaxf(a0 * sc + sh, 0.f);
  out[512 + o]  = fmaxf(a1 * sc + sh, 0.f);
  out[1024 + o] = fmaxf(a2 * sc + sh, 0.f);
  out[1536 + o] = fmaxf(a3 * sc + sh, 0.f);
}

__global__ void fc2_kernel(const float* __restrict__ h1, const float* __restrict__ W,
                           const float* __restrict__ bias, float* __restrict__ out)
{
  int i = blockIdx.x * 256 + threadIdx.x;         // 4*256
  if (i >= B_ * 256) return;
  int b = i >> 8, o = i & 255;
  const float* xr = h1 + (size_t)b * 512;
  float s = bias[o];
  for (int kk = 0; kk < 512; ++kk) s += xr[kk] * W[(size_t)kk * 256 + o];
  out[i] = s;
}

// ---------------------------------------------------------------------------
// Host side
// ---------------------------------------------------------------------------
enum {
  IN_POS = 0, IN_NRM, IN_BATCH,
  PP1_W, PP1_B, PP1_G, PP1_BE,
  PP2_W, PP2_B, PP2_G, PP2_BE,
  PP3_W, PP3_B, PP3_G, PP3_BE,
  MLPA_W, MLPA_B, MLPA_G, MLPA_BE,
  MLPB_W, MLPB_B, MLPB_G, MLPB_BE,
  MLPC_W, MLPC_B, MLPC_G, MLPC_BE,
  C1_WL, C1_BL, C1_W1, C1_B1, C1_W2, C1_B2, C1_G1, C1_BE1, C1_G2, C1_BE2,
  C2_WL, C2_BL, C2_W1, C2_B1, C2_W2, C2_B2, C2_G1, C2_BE1, C2_G2, C2_BE2,
  C3_WL, C3_BL, C3_W1, C3_B1, C3_W2, C3_B2, C3_G1, C3_BE1, C3_G2, C3_BE2,
  FC_W1, FC_B1, FC_G, FC_BE, FC_W2, FC_B2
};

extern "C" void kernel_launch(void* const* d_in, const int* in_sizes, int n_in,
                              void* d_out, int out_size, void* d_ws, size_t ws_size,
                              hipStream_t stream)
{
  (void)in_sizes; (void)n_in; (void)out_size; (void)ws_size;
  const float* pos = (const float*)d_in[IN_POS];
  const float* nrm = (const float*)d_in[IN_NRM];
  auto F = [&](int i) { return (const float*)d_in[i]; };

  // ---- workspace arena ----
  char* base = (char*)d_ws; size_t off = 0;
  auto alloc = [&](size_t n) { size_t a = (off + 255) & ~(size_t)255; off = a + n; return (void*)(base + a); };

  int*       idx   = (int*)  alloc((size_t)E_ * 4);
  float*     d2    = (float*)alloc((size_t)E_ * 4);
  float*     stats = (float*)alloc(2 * 2048 * 4);
  float*     scale = (float*)alloc(2048 * 4);
  float*     shift = (float*)alloc(2048 * 4);

  auto ah = [&](size_t n) { return (_Float16*)alloc(n * 2); };
  _Float16 *w_pp1 = ah(32*64),   *w_pp2 = ah(64*128),  *w_pp3 = ah(128*256);
  _Float16 *w_ma  = ah(32*64),   *w_mb  = ah(128*256), *w_mc  = ah((size_t)768*2048);
  _Float16 *wl1   = ah(64*384),  *w11   = ah(64*128),  *w12   = ah(64*128);
  _Float16 *wl2   = ah(256*384), *w21   = ah((size_t)256*512), *w22 = ah(64*512);
  _Float16 *wl3   = ah((size_t)768*384), *w31 = ah((size_t)768*768), *w32 = ah(64*768);

  _Float16* edge_h = ah((size_t)E_ * 128);                // pp edge features (max [E,128])
  _Float16* hpre_h = ah((size_t)E_ * 256);                // pp pre-BN (max [E,256]); aliased as x10h later
  float*    x1   = (float*)alloc((size_t)P_ * 64 * 4);
  float*    x2   = (float*)alloc((size_t)P_ * 128 * 4);
  float*    tmp64= (float*)alloc((size_t)P_ * 64 * 4);    // mlpa pre-BN
  float*    yf   = (float*)alloc((size_t)P_ * 64 * 4);
  _Float16* yh   = ah((size_t)P_ * 64);
  _Float16* x4h  = ah((size_t)P_ * 64);
  float*    x5f  = (float*)alloc((size_t)P_ * 128 * 4);
  _Float16* x5h  = ah((size_t)P_ * 128);
  float*    x6f  = (float*)alloc((size_t)P_ * 256 * 4);
  _Float16* x6h  = ah((size_t)P_ * 256);
  float*    proj = (float*)alloc((size_t)P_ * 512 * 4);   // shared: proj[P,384] / x7f[P,512]
  float*    x7f  = proj;                                  // alias (proj dead before c2-out)
  _Float16* x8h  = ah((size_t)P_ * 768);
  float*    x9f  = (float*)((char*)edge_h);               // alias edge slab (dead after pp3)
  _Float16* x9h  = ah((size_t)P_ * 768);
  _Float16* x10h = hpre_h;                                // alias hpre slab (dead after pp3)
  float*    xg   = (float*)alloc((size_t)B_ * 2048 * 4);
  float*    h1p  = (float*)alloc((size_t)B_ * 512 * 4);
  float*    h1   = (float*)alloc((size_t)B_ * 512 * 4);

  auto zero = [&](float* p, int n) {
    zero_f32<<<(n + 255) / 256, 256, 0, stream>>>(p, n);
  };
  auto cw = [&](int wi, _Float16* o, int kin, int nc, int kpad) {
    long tot = (long)kpad * nc;
    convert_w_f16<<<(int)((tot + 255) / 256), 256, 0, stream>>>(F(wi), o, kin, nc, tot);
  };
  auto gemm = [&](const _Float16* A, int lda, const _Float16* W, const float* bias,
                  float* C, int ldc, _Float16* Ch, int ldch,
                  int M, int Ka, int Nc, float* st, int accum) {
    gemm_wmma_f16<<<dim3(Nc / 64, M / 64), 128, 0, stream>>>(
        A, lda, W, bias, C, ldc, Ch, ldch, M, Ka, Nc, st, accum);
  };
  auto finalize = [&](int nc, float invM, int gi, int bei) {
    bn_finalize<<<(nc + 63) / 64, 64, 0, stream>>>(stats, nc, invM, F(gi), F(bei), scale, shift);
  };

  // ---- weight conversion (every call; deterministic) ----
  cw(PP1_W, w_pp1, 6, 64, 32);    cw(PP2_W, w_pp2, 64, 128, 64);
  cw(PP3_W, w_pp3, 128, 256, 128);
  cw(MLPA_W, w_ma, 6, 64, 32);    cw(MLPB_W, w_mb, 128, 256, 128);
  cw(MLPC_W, w_mc, 768, 2048, 768);
  cw(C1_W1, w11, 64, 128, 64);    cw(C1_W2, w12, 64, 128, 64);
  cw(C2_W1, w21, 256, 512, 256);  cw(C2_W2, w22, 64, 512, 64);
  cw(C3_W1, w31, 768, 768, 768);  cw(C3_W2, w32, 64, 768, 64);
  pack_wl_f16<<<(64 * 384 + 255) / 256, 256, 0, stream>>>(F(C1_WL), wl1, 64);
  pack_wl_f16<<<(256 * 384 + 255) / 256, 256, 0, stream>>>(F(C2_WL), wl2, 256);
  pack_wl_f16<<<(768 * 384 + 255) / 256, 256, 0, stream>>>(F(C3_WL), wl3, 768);

  // ---- kNN (shared by all graph stages) ----
  knn_kernel<<<B_ * (N_ / 256), 256, 0, stream>>>(pos, idx, d2);

  const float invE = 1.f / (float)E_, invP = 1.f / (float)P_;

  // ---- pp1: [E,6->64], BN over E, relu, max_k -> x1 [P,64]
  build_edge1<<<E_ / 256, 256, 0, stream>>>(pos, nrm, idx, edge_h);
  zero(stats, 2 * 64);
  gemm(edge_h, 32, w_pp1, F(PP1_B), nullptr, 0, hpre_h, 64, E_, 32, 64, stats, 0);
  finalize(64, invE, PP1_G, PP1_BE);
  pp_pool<<<(P_ * 64) / 256, 256, 0, stream>>>(hpre_h, scale, shift, x1, nullptr, 6, 0, 0);

  // ---- pp2: diffs [E,64->128] -> x2 [P,128]
  build_diff<<<(int)(((long)E_ * 64) / 256), 256, 0, stream>>>(x1, idx, edge_h, 6);
  zero(stats, 2 * 128);
  gemm(edge_h, 64, w_pp2, F(PP2_B), nullptr, 0, hpre_h, 128, E_, 64, 128, stats, 0);
  finalize(128, invE, PP2_G, PP2_BE);
  pp_pool<<<(P_ * 128) / 256, 256, 0, stream>>>(hpre_h, scale, shift, x2, nullptr, 7, 0, 0);

  // ---- pp3: diffs [E,128->256] -> x3 written as f16 into x8h cols [0,256)
  build_diff<<<(int)(((long)E_ * 128) / 256), 256, 0, stream>>>(x2, idx, edge_h, 7);
  zero(stats, 2 * 256);
  gemm(edge_h, 128, w_pp3, F(PP3_B), nullptr, 0, hpre_h, 256, E_, 128, 256, stats, 0);
  finalize(256, invE, PP3_G, PP3_BE);
  pp_pool<<<(P_ * 256) / 256, 256, 0, stream>>>(hpre_h, scale, shift, nullptr, x8h, 8, 768, 0);

  // ---- mlpa: [P,6->64] -> x4h
  {
    _Float16* pn = edge_h;                       // reuse slab front
    build_pn<<<P_ / 256, 256, 0, stream>>>(pos, nrm, pn);
    zero(stats, 2 * 64);
    gemm(pn, 32, w_ma, F(MLPA_B), tmp64, 64, nullptr, 0, P_, 32, 64, stats, 0);
    finalize(64, invP, MLPA_G, MLPA_BE);
    bn_apply_relu<<<(P_ * 64) / 256, 256, 0, stream>>>(tmp64, scale, shift, x4h,
                                                       64, 64, 0, (long)P_ * 64);
  }

  // ---- geoconv helper ----
  auto geoconv = [&](const _Float16* xin, int lda_in, int ci,
                     const _Float16* wlp, int bl_i,
                     const _Float16* W1h, int b1_i, const _Float16* W2h, int b2_i,
                     int g1_i, int be1_i, int g2_i, int be2_i,
                     float r2, float* outC, int co, _Float16* mirror, int ldo, int ooff) {
    gemm(xin, lda_in, wlp, F(bl_i), proj, 384, nullptr, 0, P_, ci, 384, nullptr, 0);
    zero(stats, 2 * 64);
    geo_combine<<<P_, 64, 0, stream>>>(proj, pos, idx, d2, r2, yf, stats);
    finalize(64, invP, g1_i, be1_i);
    bn_apply_relu<<<(P_ * 64) / 256, 256, 0, stream>>>(yf, scale, shift, yh,
                                                       64, 64, 0, (long)P_ * 64);
    zero(stats, 2 * co);
    gemm(xin, lda_in, W1h, F(b1_i), outC, co, nullptr, 0, P_, ci, co, nullptr, 0);
    gemm(yh, 64, W2h, F(b2_i), outC, co, nullptr, 0, P_, 64, co, stats, 1);
    finalize(co, invP, g2_i, be2_i);
    bn_apply_relu<<<(int)(((long)P_ * co) / 256), 256, 0, stream>>>(
        outC, scale, shift, mirror, co, ldo, ooff, (long)P_ * co);
  };

  // ---- c1 (r=0.15): x4 -> x5 [P,128]
  geoconv(x4h, 64, 64, wl1, C1_BL, w11, C1_B1, w12, C1_B2,
          C1_G1, C1_BE1, C1_G2, C1_BE2, 0.0225f, x5f, 128, x5h, 128, 0);

  // ---- mlpb: [P,128->256] -> x6
  zero(stats, 2 * 256);
  gemm(x5h, 128, w_mb, F(MLPB_B), x6f, 256, nullptr, 0, P_, 128, 256, stats, 0);
  finalize(256, invP, MLPB_G, MLPB_BE);
  bn_apply_relu<<<(P_ * 256) / 256, 256, 0, stream>>>(x6f, scale, shift, x6h,
                                                      256, 256, 0, (long)P_ * 256);

  // ---- c2 (r=0.3): x6 -> x7 [P,512], f16 written into x8h cols [256,768)
  geoconv(x6h, 256, 256, wl2, C2_BL, w21, C2_B1, w22, C2_B2,
          C2_G1, C2_BE1, C2_G2, C2_BE2, 0.09f, x7f, 512, x8h, 768, 256);

  // ---- c3 (r=0.6): x8 [P,768] -> x9 [P,768]
  geoconv(x8h, 768, 768, wl3, C3_BL, w31, C3_B1, w32, C3_B2,
          C3_G1, C3_BE1, C3_G2, C3_BE2, 0.36f, x9f, 768, x9h, 768, 0);

  // ---- mlpc: [P,768->2048], BN, relu, global max over N -> xg [4,2048]
  zero(stats, 2 * 2048);
  gemm(x9h, 768, w_mc, F(MLPC_B), nullptr, 0, x10h, 2048, P_, 768, 2048, stats, 0);
  finalize(2048, invP, MLPC_G, MLPC_BE);
  maxpool_bn<<<(B_ * 2048) / 256, 256, 0, stream>>>(x10h, scale, shift, xg);

  // ---- fc head (tiny, f32 VALU) ----
  fc1_kernel<<<(B_ * 512) / 256, 256, 0, stream>>>(xg, F(FC_W1), F(FC_B1), h1p);
  fc_bn<<<2, 256, 0, stream>>>(h1p, F(FC_G), F(FC_BE), h1);
  fc2_kernel<<<(B_ * 256) / 256, 256, 0, stream>>>(h1, F(FC_W2), F(FC_B2), (float*)d_out);
}
